// RecurrentClassifier_4269197492582
// MI455X (gfx1250) — matrix operand order repaired and checked by reading, not compiled, not run on hardware
//
#include <hip/hip_runtime.h>

typedef __bf16 bf16;
typedef __attribute__((ext_vector_type(16))) __bf16 v16bf;
typedef __attribute__((ext_vector_type(8)))  __bf16 v8bf;
typedef __attribute__((ext_vector_type(8)))  float  v8f;
typedef int v4i __attribute__((vector_size(16)));

#define B_   256
#define T_   32
#define D_   4096
#define L1_  4096
#define NC_  10
#define BT_  (B_ * T_)

#define BETA 0.9f
#define THR  1.0f

// Block tile geometry for the WMMA GEMMs
#define BM 128
#define BN 128
#define BK 64

// ---------------------------------------------------------------------------
// CDNA5 async global->LDS copy (ASYNCcnt-tracked DMA), with safe fallback.
// This clang declares the b128 variant as (v4i AS1*, v4i AS3*, imm, imm).
// ---------------------------------------------------------------------------
#if defined(__has_builtin)
#if __has_builtin(__builtin_amdgcn_global_load_async_to_lds_b128)
#define HAVE_ASYNC_LDS 1
#endif
#endif

#ifdef HAVE_ASYNC_LDS
typedef __attribute__((address_space(1))) v4i* gv4i_p;
typedef __attribute__((address_space(3))) v4i* lv4i_p;
__device__ __forceinline__ void async_cp16(const void* g, void* l) {
    __builtin_amdgcn_global_load_async_to_lds_b128(
        (gv4i_p)(uintptr_t)g, (lv4i_p)(unsigned)(uintptr_t)l, 0, 0);
}
#if __has_builtin(__builtin_amdgcn_s_wait_asynccnt)
#define WAIT_ASYNC(n) __builtin_amdgcn_s_wait_asynccnt(n)
#else
#define WAIT_ASYNC(n) asm volatile("s_wait_asynccnt %0" ::"n"(n))
#endif
#else
#define WAIT_ASYNC(n) ((void)0)
#endif

// ---------------------------------------------------------------------------
// Fragment loaders (LDS tile is row-major [rows][BK] bf16, 16B aligned rows).
// A-matrix 16x32 bf16 layout (ISA 7.12.2): lane l holds row M=l%16;
//   lane-half kg=l/16: halves {kg*8..kg*8+7} then {16+kg*8..16+kg*8+7}.
// B-matrix 32x16 layout: lane l holds column N=l%16 (== row of W, since we
//   compute X @ W^T); same K order as A: {kg*8..kg*8+7} then {16+kg*8..16+kg*8+7}.
// ---------------------------------------------------------------------------
__device__ __forceinline__ v16bf load_fragA(const bf16* s, int row, int kb, int lane) {
    const int kg = (lane >> 4) & 1;
    const bf16* p = s + row * BK + kb + kg * 8;
    v8bf lo = *(const v8bf*)(p);
    v8bf hi = *(const v8bf*)(p + 16);
    return __builtin_shufflevector(lo, hi, 0,1,2,3,4,5,6,7,8,9,10,11,12,13,14,15);
}

__device__ __forceinline__ v16bf load_fragB(const bf16* s, int row, int kb, int lane) {
    const int kg = (lane >> 4) & 1;
    const bf16* p = s + row * BK + kb + kg * 8;
    v8bf lo = *(const v8bf*)(p);
    v8bf hi = *(const v8bf*)(p + 16);
    return __builtin_shufflevector(lo, hi, 0,1,2,3,4,5,6,7,8,9,10,11,12,13,14,15);
}

// Stage loader: 128x64 bf16 tiles for A and B (8 x 16B chunks per thread).
__device__ __forceinline__ void load_stage(bf16* dA, bf16* dB,
                                           const bf16* __restrict__ A,
                                           const bf16* __restrict__ Bm, int bM,
                                           int bN, int k0, int tid) {
#pragma unroll
    for (int i = 0; i < 4; ++i) {
        const int c   = tid + i * 256;
        const int row = c >> 3;
        const int kc  = (c & 7) << 3;
#ifdef HAVE_ASYNC_LDS
        async_cp16(&A[(size_t)(bM + row) * 4096 + k0 + kc], &dA[row * BK + kc]);
        async_cp16(&Bm[(size_t)(bN + row) * 4096 + k0 + kc], &dB[row * BK + kc]);
#else
        *(uint4*)(&dA[row * BK + kc]) =
            *(const uint4*)(&A[(size_t)(bM + row) * 4096 + k0 + kc]);
        *(uint4*)(&dB[row * BK + kc]) =
            *(const uint4*)(&Bm[(size_t)(bN + row) * 4096 + k0 + kc]);
#endif
    }
}

// ---------------------------------------------------------------------------
// Unified WMMA GEMM: C[M,4096] = A[M,4096] @ Bm[4096,4096]^T  (bf16 -> f32)
// MODE 0: precompute cur1 = X @ W1^T + b1           (M = B*T)
// MODE 1: recurrent step: rec = spk_old @ Wr1^T; then fused RLeaky update:
//         mem1 = BETA*mem1 + cur1[:,t,:] + rec + br1 - spk_old*THR
//         spk_new = (mem1 - THR) > 0                 (M = B)
// Double-buffered LDS stages filled by async global->LDS DMA.
// ---------------------------------------------------------------------------
template <int MODE>
__global__ __launch_bounds__(256) void wmma_gemm_kernel(
    const bf16* __restrict__ A, const bf16* __restrict__ Bm,
    const float* __restrict__ bias,       // MODE0: b1, MODE1: br1
    float* __restrict__ outCur,           // MODE0 output
    const float* __restrict__ cur1,       // MODE1 input [B*T, L1]
    float* __restrict__ mem1,             // MODE1 in/out [B, L1]
    const bf16* __restrict__ spkOld,      // MODE1 [B, L1]
    bf16* __restrict__ spkNew,            // MODE1 [B, L1]
    int t) {
    __shared__ __attribute__((aligned(16))) bf16 sA[2][BM * BK];
    __shared__ __attribute__((aligned(16))) bf16 sB[2][BN * BK];

    const int tid  = threadIdx.x;
    const int lane = tid & 31;
    const int wave = tid >> 5;          // 8 waves
    const int wm   = wave >> 1;         // 0..3 -> 32-row strips
    const int wn   = wave & 1;          // 0..1 -> 64-col strips
    const int bM   = blockIdx.y * BM;
    const int bN   = blockIdx.x * BN;

    v8f acc[2][4];
#pragma unroll
    for (int m = 0; m < 2; ++m)
#pragma unroll
        for (int n = 0; n < 4; ++n) acc[m][n] = (v8f){0.f,0.f,0.f,0.f,0.f,0.f,0.f,0.f};

    constexpr int NIT = 4096 / BK;  // 64 K-stages
    load_stage(sA[0], sB[0], A, Bm, bM, bN, 0, tid);

    for (int it = 0; it < NIT; ++it) {
        const bf16* cA = sA[it & 1];
        const bf16* cB = sB[it & 1];
        if (it + 1 < NIT) {
            // Prefetch next stage into the other buffer (its readers finished
            // at the trailing barrier of the previous iteration).
            load_stage(sA[(it + 1) & 1], sB[(it + 1) & 1], A, Bm, bM, bN,
                       (it + 1) * BK, tid);
            WAIT_ASYNC(8);   // leave next stage's 8 ops in flight; stage `it` landed
        } else {
            WAIT_ASYNC(0);
        }
        __syncthreads();

#pragma unroll
        for (int kb = 0; kb < BK; kb += 32) {
            v16bf aF[2], bF[4];
            const int r0 = wm * 32;
            const int c0 = wn * 64;
            aF[0] = load_fragA(cA, r0 + (lane & 15), kb, lane);
            aF[1] = load_fragA(cA, r0 + 16 + (lane & 15), kb, lane);
#pragma unroll
            for (int n = 0; n < 4; ++n)
                bF[n] = load_fragB(cB, c0 + n * 16 + (lane & 15), kb, lane);
#pragma unroll
            for (int m = 0; m < 2; ++m)
#pragma unroll
                for (int n = 0; n < 4; ++n)
                    acc[m][n] = __builtin_amdgcn_wmma_f32_16x16x32_bf16(
                        false, aF[m], false, bF[n], (short)0, acc[m][n], false, false);
        }
        __syncthreads();  // all waves done reading before this buffer is refilled
    }

    // Epilogue. C layout: v8f element r, lane l -> row += (l<16 ? r : r+8), col = l%16.
    const int laneN = lane & 15;
    const int rOff  = (lane & 16) ? 8 : 0;
#pragma unroll
    for (int m = 0; m < 2; ++m)
#pragma unroll
        for (int n = 0; n < 4; ++n)
#pragma unroll
            for (int r = 0; r < 8; ++r) {
                const int row = bM + wm * 32 + m * 16 + rOff + r;
                const int col = bN + wn * 64 + n * 16 + laneN;
                const float v = acc[m][n][r];
                if (MODE == 0) {
                    outCur[(size_t)row * L1_ + col] = v + bias[col];
                } else {
                    const size_t idx  = (size_t)row * L1_ + col;
                    const float olds  = (float)spkOld[idx];
                    const float mm    = BETA * mem1[idx] +
                                        cur1[((size_t)row * T_ + t) * L1_ + col] +
                                        v + bias[col] - olds * THR;
                    mem1[idx]   = mm;
                    spkNew[idx] = (bf16)(((mm - THR) > 0.0f) ? 1.0f : 0.0f);
                }
            }
}

// ---------------------------------------------------------------------------
// Prep: fp32 -> bf16 conversions + state init (runs every call; deterministic)
// ---------------------------------------------------------------------------
__global__ void prep_kernel(const float* __restrict__ x, const float* __restrict__ W1,
                            const float* __restrict__ Wr1, bf16* __restrict__ Xbf,
                            bf16* __restrict__ W1bf, bf16* __restrict__ Wr1bf,
                            float* __restrict__ mem1, bf16* __restrict__ spk1a,
                            float* __restrict__ mem2, float* __restrict__ spk2) {
    const size_t tid    = (size_t)blockIdx.x * blockDim.x + threadIdx.x;
    const size_t stride = (size_t)gridDim.x * blockDim.x;
    for (size_t i = tid; i < (size_t)BT_ * D_; i += stride) Xbf[i] = (bf16)x[i];
    for (size_t i = tid; i < (size_t)L1_ * D_; i += stride) W1bf[i] = (bf16)W1[i];
    for (size_t i = tid; i < (size_t)L1_ * L1_; i += stride) Wr1bf[i] = (bf16)Wr1[i];
    for (size_t i = tid; i < (size_t)B_ * L1_; i += stride) {
        mem1[i]  = 0.0f;
        spk1a[i] = (bf16)0.0f;
    }
    for (size_t i = tid; i < (size_t)B_ * NC_; i += stride) {
        mem2[i] = 0.0f;
        spk2[i] = 0.0f;
    }
}

// ---------------------------------------------------------------------------
// Layer 2 (tiny, N=10): one block per batch row; fused RLeaky + output write.
// ---------------------------------------------------------------------------
__global__ __launch_bounds__(256) void step2_kernel(
    const bf16* __restrict__ spk1, const float* __restrict__ W2,
    const float* __restrict__ b2, const float* __restrict__ Wr2,
    const float* __restrict__ br2, float* __restrict__ mem2,
    float* __restrict__ spk2, float* __restrict__ out, int t) {
    const int b   = blockIdx.x;
    const int tid = threadIdx.x;

    float accv[NC_];
#pragma unroll
    for (int c = 0; c < NC_; ++c) accv[c] = 0.0f;

    for (int k = tid; k < L1_; k += 256) {
        const float s = (float)spk1[(size_t)b * L1_ + k];
        if (s != 0.0f) {
#pragma unroll
            for (int c = 0; c < NC_; ++c) accv[c] += W2[(size_t)c * L1_ + k];
        }
    }

    __shared__ float red[256];
    __shared__ float s_cur[NC_], s_spk[NC_], s_mem[NC_];
    for (int c = 0; c < NC_; ++c) {
        red[tid] = accv[c];
        __syncthreads();
        for (int off = 128; off > 0; off >>= 1) {
            if (tid < off) red[tid] += red[tid + off];
            __syncthreads();
        }
        if (tid == 0) s_cur[c] = red[0] + b2[c];
        __syncthreads();
    }

    if (tid < NC_) {
        s_spk[tid] = spk2[b * NC_ + tid];
        s_mem[tid] = mem2[b * NC_ + tid];
    }
    __syncthreads();
    if (tid < NC_) {
        float rec = 0.0f;
#pragma unroll
        for (int j = 0; j < NC_; ++j) rec += s_spk[j] * Wr2[tid * NC_ + j];
        const float mm = BETA * s_mem[tid] + s_cur[tid] + rec + br2[tid] - s_spk[tid] * THR;
        const float sp = ((mm - THR) > 0.0f) ? 1.0f : 0.0f;
        mem2[b * NC_ + tid] = mm;
        spk2[b * NC_ + tid] = sp;
        out[(size_t)t * B_ * NC_ + (size_t)b * NC_ + tid]                         = sp;  // spk_rec
        out[(size_t)T_ * B_ * NC_ + (size_t)t * B_ * NC_ + (size_t)b * NC_ + tid] = mm;  // mem_rec
    }
}

// ---------------------------------------------------------------------------
extern "C" void kernel_launch(void* const* d_in, const int* in_sizes, int n_in,
                              void* d_out, int out_size, void* d_ws, size_t ws_size,
                              hipStream_t stream) {
    (void)in_sizes; (void)n_in; (void)out_size; (void)ws_size;
    const float* x   = (const float*)d_in[0];
    const float* W1  = (const float*)d_in[1];
    const float* b1  = (const float*)d_in[2];
    const float* Wr1 = (const float*)d_in[3];
    const float* br1 = (const float*)d_in[4];
    const float* W2  = (const float*)d_in[5];
    const float* b2  = (const float*)d_in[6];
    const float* Wr2 = (const float*)d_in[7];
    const float* br2 = (const float*)d_in[8];
    float* out = (float*)d_out;

    char* ws   = (char*)d_ws;
    size_t off = 0;
    auto salloc = [&](size_t bytes) -> char* {
        char* p = ws + off;
        off += (bytes + 255) & ~(size_t)255;
        return p;
    };
    bf16*  Xbf   = (bf16*)salloc((size_t)BT_ * D_ * sizeof(bf16));   //  64 MB
    bf16*  W1bf  = (bf16*)salloc((size_t)L1_ * D_ * sizeof(bf16));   //  32 MB
    bf16*  Wr1bf = (bf16*)salloc((size_t)L1_ * L1_ * sizeof(bf16));  //  32 MB
    float* cur1  = (float*)salloc((size_t)BT_ * L1_ * sizeof(float));// 128 MB
    float* mem1  = (float*)salloc((size_t)B_ * L1_ * sizeof(float)); //   4 MB
    bf16*  spkA  = (bf16*)salloc((size_t)B_ * L1_ * sizeof(bf16));   //   2 MB
    bf16*  spkB  = (bf16*)salloc((size_t)B_ * L1_ * sizeof(bf16));   //   2 MB
    float* mem2  = (float*)salloc((size_t)B_ * NC_ * sizeof(float));
    float* spk2  = (float*)salloc((size_t)B_ * NC_ * sizeof(float));

    // 1) bf16 conversions + zero state
    prep_kernel<<<2048, 256, 0, stream>>>(x, W1, Wr1, Xbf, W1bf, Wr1bf, mem1, spkA,
                                          mem2, spk2);

    // 2) Precompute cur1 = X @ W1^T + b1  (8192 x 4096 x 4096)
    dim3 g0(L1_ / BN, BT_ / BM);  // (32, 64)
    wmma_gemm_kernel<0><<<g0, 256, 0, stream>>>(Xbf, W1bf, b1, cur1, nullptr, nullptr,
                                                nullptr, nullptr, 0);

    // 3) Sequential time loop: recurrent WMMA GEMM fused with RLeaky, then layer 2.
    dim3 g1(L1_ / BN, B_ / BM);  // (32, 2)
    bf16* bufs[2] = {spkA, spkB};
    for (int t = 0; t < T_; ++t) {
        bf16* so = bufs[t & 1];
        bf16* sn = bufs[(t + 1) & 1];
        wmma_gemm_kernel<1><<<g1, 256, 0, stream>>>(so, Wr1bf, br1, nullptr, cur1,
                                                    mem1, so, sn, t);
        step2_kernel<<<B_, 256, 0, stream>>>(sn, W2, b2, Wr2, br2, mem2, spk2, out, t);
    }
}
